// CustomChemicalAttention_18614388261049
// MI455X (gfx1250) — compile-verified, matrix operand order checked by reading, not run on hardware
//
#include <hip/hip_runtime.h>
#include <hip/hip_bf16.h>
#include <math.h>

// ---------------------------------------------------------------------------
// MHA forward for MI455X (gfx1250, wave32, WMMA).
// Roofline: ~52 GFLOP compute, but 256MB of mandatory attn_probs writes
// -> HBM-write bound (~11us floor @ 23.3 TB/s). All GEMMs use
// v_wmma_f32_16x16x32_f16 (f16 in, f32 accumulate) for accuracy + speed.
// All inputs/weights (~64MB) fit in the 192MB L2, so redundant K/V re-reads
// across attention blocks are L2 hits, not HBM traffic.
// ---------------------------------------------------------------------------

#define EMBED 1024
#define HEADS 16
#define HD 64
#define BQ 64        // q-rows per attention workgroup (score tile = 256KB LDS)

typedef __attribute__((ext_vector_type(16))) _Float16 v16h;
typedef __attribute__((ext_vector_type(8)))  _Float16 v8h;
typedef __attribute__((ext_vector_type(4)))  _Float16 v4h;
typedef __attribute__((ext_vector_type(8)))  float    v8f;
typedef __attribute__((ext_vector_type(4)))  float    v4f;

// Load one 16x32 f16 A/B fragment for the current lane.
// ISA layout (CDNA5 ISA): lane<16 holds K = {k0a..k0a+7, k0b..k0b+7} with
// k0a = base + half*8, k0b = base + 16 + half*8. Both groups are contiguous
// in memory -> two 16-byte loads.
__device__ inline v16h frag_ld(const _Float16* p, int k0a, int k0b) {
  v8h lo = *reinterpret_cast<const v8h*>(p + k0a);
  v8h hi = *reinterpret_cast<const v8h*>(p + k0b);
  v16h r;
#pragma unroll
  for (int i = 0; i < 8; ++i) { r[i] = lo[i]; r[i + 8] = hi[i]; }
  return r;
}

__device__ inline v8f wmma_f16(v16h a, v16h b, v8f c) {
  // (neg_a, A, neg_b, B, c_mod, C, reuse_a, reuse_b)
  return __builtin_amdgcn_wmma_f32_16x16x32_f16(false, a, false, b,
                                                (short)0, c, false, false);
}

// ---------------------------------------------------------------------------
// Kernel 1: Y = (X @ W + bias) * scale, X/W f32 in global, Y f16.
// WRITE_VT (compile-time): also write YT[(b*1024 + N)*1024 + s]
// (head-transposed V copy so the PV GEMM's B-fragments are contiguous).
// Tile: 128x128x32, 256 threads = 8 waves, wave = 2x4 WMMA tiles (32x64).
// ---------------------------------------------------------------------------
template <bool WRITE_VT>
__global__ __launch_bounds__(256) void gemm_proj_kernel(
    const float* __restrict__ X, const float* __restrict__ W,
    const float* __restrict__ bias, float scale,
    _Float16* __restrict__ Y, _Float16* __restrict__ YT) {
  __shared__ alignas(16) _Float16 As[128][40];   // row-major [m][k], padded
  __shared__ alignas(16) _Float16 Bs[128][40];   // n-major  [n][k], padded

  const int tid  = threadIdx.x;
  const int lane = tid & 31, w = tid >> 5;
  const int half = lane >> 4, lr = lane & 15;
  const int wm0  = (w & 3) * 32, wn0 = (w >> 2) * 64;
  const int m0   = blockIdx.y * 128, n0 = blockIdx.x * 128;
  const int k0a  = half * 8, k0b = 16 + half * 8;

  v8f acc[2][4];
#pragma unroll
  for (int mt = 0; mt < 2; ++mt)
#pragma unroll
    for (int nt = 0; nt < 4; ++nt) acc[mt][nt] = (v8f){};

  for (int kk = 0; kk < EMBED; kk += 32) {
    // Stage A: 128x32 f32 -> f16 LDS (row-major).
#pragma unroll
    for (int i = 0; i < 4; ++i) {
      int idx = tid + 256 * i;
      int row = idx >> 3, c4 = (idx & 7) * 4;
      v4f x = *reinterpret_cast<const v4f*>(X + (size_t)(m0 + row) * EMBED + kk + c4);
      v4h hv;
#pragma unroll
      for (int j = 0; j < 4; ++j) hv[j] = (_Float16)x[j];
      *reinterpret_cast<v4h*>(&As[row][c4]) = hv;
    }
    // Stage B: 32x128 f32 -> transposed f16 LDS (n-major so frag loads are
    // contiguous in K).
#pragma unroll
    for (int i = 0; i < 4; ++i) {
      int idx = tid + 256 * i;
      int kr = idx >> 5, c4 = (idx & 31) * 4;
      v4f wv = *reinterpret_cast<const v4f*>(W + (size_t)(kk + kr) * EMBED + n0 + c4);
#pragma unroll
      for (int j = 0; j < 4; ++j) Bs[c4 + j][kr] = (_Float16)wv[j];
    }
    __syncthreads();

    v16h af[2], bf[4];
#pragma unroll
    for (int mt = 0; mt < 2; ++mt) af[mt] = frag_ld(&As[wm0 + mt * 16 + lr][0], k0a, k0b);
#pragma unroll
    for (int nt = 0; nt < 4; ++nt) bf[nt] = frag_ld(&Bs[wn0 + nt * 16 + lr][0], k0a, k0b);
#pragma unroll
    for (int mt = 0; mt < 2; ++mt)
#pragma unroll
      for (int nt = 0; nt < 4; ++nt)
        acc[mt][nt] = wmma_f16(af[mt], bf[nt], acc[mt][nt]);
    __syncthreads();
  }

  // Epilogue. C/D layout: VGPR r -> M = half*8 + r, N = lane&15.
#pragma unroll
  for (int mt = 0; mt < 2; ++mt) {
#pragma unroll
    for (int nt = 0; nt < 4; ++nt) {
      int N = n0 + wn0 + nt * 16 + lr;
      float bv = bias[N];
#pragma unroll
      for (int r = 0; r < 8; ++r) {
        int M = m0 + wm0 + mt * 16 + half * 8 + r;
        _Float16 hv = (_Float16)((acc[mt][nt][r] + bv) * scale);
        Y[(size_t)M * EMBED + N] = hv;
        if constexpr (WRITE_VT) {  // V: [b, h*64+d, s] layout, branch-free
          YT[((size_t)((M >> 10) * EMBED + N)) * 1024 + (M & 1023)] = hv;
        }
      }
    }
  }
}

// ---------------------------------------------------------------------------
// Kernel 2: attention core for one (b, h, 64 q-rows) block.
// Phase 1: S = Q Kt (WMMA, frags loaded straight from global f16) + mask.
// Phase 2: row softmax (wave32 shfl reductions); probs -> global (f32).
// Phase 3: ctx = P V (WMMA; P converted f32->f16 from LDS via b128 loads).
// Dynamic LDS: BQ*1024 f32 = 256 KB (CDNA5: up to 320 KB per workgroup).
// ---------------------------------------------------------------------------
__global__ __launch_bounds__(256) void attn_kernel(
    const _Float16* __restrict__ qh, const _Float16* __restrict__ kh,
    const _Float16* __restrict__ vT, const unsigned char* __restrict__ mask,
    float* __restrict__ probs, _Float16* __restrict__ ctxh) {
  extern __shared__ float sc[];  // [BQ][1024] scores -> probs
  const int tid  = threadIdx.x;
  const int lane = tid & 31, w = tid >> 5;
  const int half = lane >> 4, lr = lane & 15;
  const int q0 = blockIdx.x * BQ, h = blockIdx.y, b = blockIdx.z;
  const float NEG_INF = -__builtin_huge_valf();

  // ---- Phase 1: scores[64][1024] ----
  {
    const int mt = w >> 1;                       // waves pair up on m-tiles
    v16h aq[2];
    const _Float16* qp = qh + ((size_t)(b * 1024 + q0 + mt * 16 + lr)) * EMBED + h * HD;
#pragma unroll
    for (int t = 0; t < 2; ++t)
      aq[t] = frag_ld(qp, t * 32 + half * 8, t * 32 + 16 + half * 8);

    for (int j = 0; j < 32; ++j) {
      int nt = (w & 1) + 2 * j;                  // odd/even n-tiles per wave
      const _Float16* kp = kh + ((size_t)(b * 1024 + nt * 16 + lr)) * EMBED + h * HD;
      v8f s = (v8f){};
#pragma unroll
      for (int t = 0; t < 2; ++t) {
        v16h bk = frag_ld(kp, t * 32 + half * 8, t * 32 + 16 + half * 8);
        s = wmma_f16(aq[t], bk, s);
      }
      int col = nt * 16 + lr;
      bool msk = mask[b * 1024 + col] != 0;      // jnp bool -> 1 byte
#pragma unroll
      for (int r = 0; r < 8; ++r) {
        int rowl = mt * 16 + half * 8 + r;
        sc[rowl * 1024 + col] = msk ? NEG_INF : s[r];
      }
    }
  }
  __syncthreads();

  // ---- Phase 2: softmax over each of the 64 rows ----
  for (int rr = 0; rr < 8; ++rr) {
    int row = w * 8 + rr;
    float* rp = sc + row * 1024;
    float m = NEG_INF;
    for (int j = 0; j < 32; ++j) m = fmaxf(m, rp[lane + 32 * j]);
#pragma unroll
    for (int off = 16; off >= 1; off >>= 1) m = fmaxf(m, __shfl_xor(m, off, 32));
    float sum = 0.f;
    for (int j = 0; j < 32; ++j) {
      int idx = lane + 32 * j;
      float e = __expf(rp[idx] - m);             // exp(-inf) = 0 for masked
      rp[idx] = e;
      sum += e;
    }
#pragma unroll
    for (int off = 16; off >= 1; off >>= 1) sum += __shfl_xor(sum, off, 32);
    float inv = 1.f / sum;
    size_t base = ((size_t)(b * HEADS + h)) * 1024 * 1024 + (size_t)(q0 + row) * 1024;
    for (int j = 0; j < 32; ++j) {
      int idx = lane + 32 * j;                   // coalesced 128B per j
      float p = rp[idx] * inv;
      rp[idx] = p;
      probs[base + idx] = p;
    }
  }
  __syncthreads();

  // ---- Phase 3: ctx[64][64] = P[64][1024] x V[1024][64] ----
  {
    const int mt = w >> 1, ntb = (w & 1) * 2;
    v8f c0 = (v8f){}, c1 = (v8f){};
    const float* rp = sc + (mt * 16 + lr) * 1024;
    const _Float16* vp0 = vT + ((size_t)(b * 1024 + h * HD + ntb * 16 + lr)) * 1024;
    const _Float16* vp1 = vp0 + (size_t)16 * 1024;
    for (int ks = 0; ks < 1024; ks += 32) {
      // P fragment: explicit 16B LDS loads (ds_load_b128), then f32->f16.
      v4f f0 = *reinterpret_cast<const v4f*>(rp + ks + half * 8);
      v4f f1 = *reinterpret_cast<const v4f*>(rp + ks + half * 8 + 4);
      v4f f2 = *reinterpret_cast<const v4f*>(rp + ks + 16 + half * 8);
      v4f f3 = *reinterpret_cast<const v4f*>(rp + ks + 16 + half * 8 + 4);
      v16h pa;
#pragma unroll
      for (int i = 0; i < 4; ++i) {
        pa[i]      = (_Float16)f0[i];
        pa[i + 4]  = (_Float16)f1[i];
        pa[i + 8]  = (_Float16)f2[i];
        pa[i + 12] = (_Float16)f3[i];
      }
      v16h b0 = frag_ld(vp0, ks + half * 8, ks + 16 + half * 8);
      v16h b1 = frag_ld(vp1, ks + half * 8, ks + 16 + half * 8);
      c0 = wmma_f16(pa, b0, c0);
      c1 = wmma_f16(pa, b1, c1);
    }
#pragma unroll
    for (int r = 0; r < 8; ++r) {
      int srow = q0 + mt * 16 + half * 8 + r;
      size_t rowbase = ((size_t)(b * 1024 + srow)) * EMBED + h * HD;
      ctxh[rowbase + ntb * 16 + lr]       = (_Float16)c0[r];
      ctxh[rowbase + (ntb + 1) * 16 + lr] = (_Float16)c1[r];
    }
  }
}

// ---------------------------------------------------------------------------
// Kernel 3: attn_output = ctx(f16) @ Wo(f32) + bo -> f32.
// ---------------------------------------------------------------------------
__global__ __launch_bounds__(256) void gemm_out_kernel(
    const _Float16* __restrict__ A, const float* __restrict__ W,
    const float* __restrict__ bias, float* __restrict__ Y) {
  __shared__ alignas(16) _Float16 As[128][40];
  __shared__ alignas(16) _Float16 Bs[128][40];

  const int tid  = threadIdx.x;
  const int lane = tid & 31, w = tid >> 5;
  const int half = lane >> 4, lr = lane & 15;
  const int wm0  = (w & 3) * 32, wn0 = (w >> 2) * 64;
  const int m0   = blockIdx.y * 128, n0 = blockIdx.x * 128;
  const int k0a  = half * 8, k0b = 16 + half * 8;

  v8f acc[2][4];
#pragma unroll
  for (int mt = 0; mt < 2; ++mt)
#pragma unroll
    for (int nt = 0; nt < 4; ++nt) acc[mt][nt] = (v8f){};

  for (int kk = 0; kk < EMBED; kk += 32) {
    // Stage A: already f16, straight 16B copies.
#pragma unroll
    for (int i = 0; i < 2; ++i) {
      int idx = tid + 256 * i;
      int row = idx >> 2, c8 = (idx & 3) * 8;
      *reinterpret_cast<v8h*>(&As[row][c8]) =
          *reinterpret_cast<const v8h*>(A + (size_t)(m0 + row) * EMBED + kk + c8);
    }
    // Stage B: f32 -> f16 transpose.
#pragma unroll
    for (int i = 0; i < 4; ++i) {
      int idx = tid + 256 * i;
      int kr = idx >> 5, c4 = (idx & 31) * 4;
      v4f wv = *reinterpret_cast<const v4f*>(W + (size_t)(kk + kr) * EMBED + n0 + c4);
#pragma unroll
      for (int j = 0; j < 4; ++j) Bs[c4 + j][kr] = (_Float16)wv[j];
    }
    __syncthreads();

    v16h af[2], bf[4];
#pragma unroll
    for (int mt = 0; mt < 2; ++mt) af[mt] = frag_ld(&As[wm0 + mt * 16 + lr][0], k0a, k0b);
#pragma unroll
    for (int nt = 0; nt < 4; ++nt) bf[nt] = frag_ld(&Bs[wn0 + nt * 16 + lr][0], k0a, k0b);
#pragma unroll
    for (int mt = 0; mt < 2; ++mt)
#pragma unroll
      for (int nt = 0; nt < 4; ++nt)
        acc[mt][nt] = wmma_f16(af[mt], bf[nt], acc[mt][nt]);
    __syncthreads();
  }

#pragma unroll
  for (int mt = 0; mt < 2; ++mt) {
#pragma unroll
    for (int nt = 0; nt < 4; ++nt) {
      int N = n0 + wn0 + nt * 16 + lr;
      float bv = bias[N];
#pragma unroll
      for (int r = 0; r < 8; ++r) {
        int M = m0 + wm0 + mt * 16 + half * 8 + r;
        Y[(size_t)M * EMBED + N] = acc[mt][nt][r] + bv;
      }
    }
  }
}

// ---------------------------------------------------------------------------
// Host launcher.
// Inputs: 0 query, 1 key, 2 value, 3 key_padding_mask (bool/u8),
//         4 Wq, 5 bq, 6 Wk, 7 bk, 8 Wv, 9 bv, 10 Wo, 11 bo.
// d_out: attn_output [4,1024,1024] f32 (4M) ++ attn_probs [4,16,1024,1024]
// f32 (64M). Workspace: 5 x 8MB f16 buffers (q, k, v, vT, ctx) = 40MB.
// ---------------------------------------------------------------------------
extern "C" void kernel_launch(void* const* d_in, const int* in_sizes, int n_in,
                              void* d_out, int out_size, void* d_ws, size_t ws_size,
                              hipStream_t stream) {
  (void)in_sizes; (void)n_in; (void)out_size; (void)ws_size;

  const float* query = (const float*)d_in[0];
  const float* key   = (const float*)d_in[1];
  const float* value = (const float*)d_in[2];
  const unsigned char* mask = (const unsigned char*)d_in[3];
  const float* Wq = (const float*)d_in[4];
  const float* bq = (const float*)d_in[5];
  const float* Wk = (const float*)d_in[6];
  const float* bk = (const float*)d_in[7];
  const float* Wv = (const float*)d_in[8];
  const float* bv = (const float*)d_in[9];
  const float* Wo = (const float*)d_in[10];
  const float* bo = (const float*)d_in[11];

  float* out   = (float*)d_out;
  float* probs = out + (size_t)4 * 1024 * 1024;

  const size_t MAT = (size_t)4096 * 1024;  // elements per f16 matrix
  _Float16* qh  = (_Float16*)d_ws;
  _Float16* kh  = qh + MAT;
  _Float16* vh  = kh + MAT;
  _Float16* vT  = vh + MAT;
  _Float16* ctx = vT + MAT;

  const float SCALING = 0.125f;  // 64^-0.5 (folded into Q projection)

  dim3 gproj(8, 32);  // N/128 x M/128
  gemm_proj_kernel<false><<<gproj, 256, 0, stream>>>(query, Wq, bq, SCALING, qh, nullptr);
  gemm_proj_kernel<false><<<gproj, 256, 0, stream>>>(key,   Wk, bk, 1.0f,    kh, nullptr);
  gemm_proj_kernel<true ><<<gproj, 256, 0, stream>>>(value, Wv, bv, 1.0f,    vh, vT);

  const int smem = BQ * 1024 * sizeof(float);  // 256 KB (<= 320 KB on CDNA5)
  (void)hipFuncSetAttribute((const void*)attn_kernel,
                            hipFuncAttributeMaxDynamicSharedMemorySize, smem);
  attn_kernel<<<dim3(16, HEADS, 4), 256, smem, stream>>>(qh, kh, vT, mask, probs, ctx);

  gemm_out_kernel<<<gproj, 256, 0, stream>>>(ctx, Wo, bo, out);
}
